// Gate_50946902065664
// MI455X (gfx1250) — matrix operand order repaired and checked by reading, not compiled, not run on hardware
//
#include <hip/hip_runtime.h>
#include <stdint.h>

// ---------------- problem constants (from reference) ----------------
#define D_DIM 7168        // hidden dim (K)
#define E_DIM 256         // experts (N of the gemm)
#define NGRP  8
#define TOPG  4
#define TOPK  8
#define ROUTE_SCALE 2.5f

// ---------------- tiling ----------------
#define ROWS_PER_BLOCK 64          // four 16-row WMMA M-tiles (amortize weight L2 traffic)
#define MTILES (ROWS_PER_BLOCK / 16)
#define KC 128                     // K chunk staged in LDS per iteration
#define LDS_STRIDE (KC + 8)        // bf16 elems; 272B rows: 16B aligned, 4-bank skew
#define PF ((ROWS_PER_BLOCK * KC / 4) / 256)   // f32x4 prefetch regs per thread (= 8)

typedef __attribute__((ext_vector_type(16))) __bf16         v16bf;
typedef __attribute__((ext_vector_type(8)))  float          v8f;
typedef __attribute__((ext_vector_type(4)))  float          f32x4;
typedef __attribute__((ext_vector_type(4)))  unsigned short u16x4;
typedef __attribute__((ext_vector_type(8)))  unsigned short u16x8;
typedef __attribute__((ext_vector_type(16))) unsigned short u16x16;

__device__ __forceinline__ unsigned short f32_to_bf16_rne(float f) {
    union { float f; uint32_t u; } c; c.f = f;
    uint32_t u = c.u;
    u += 0x7FFFu + ((u >> 16) & 1u);   // round-to-nearest-even
    return (unsigned short)(u >> 16);
}

__device__ __forceinline__ v16bf combine16(u16x8 lo, u16x8 hi) {
    u16x16 c;
#pragma unroll
    for (int i = 0; i < 8; ++i) { c[i] = lo[i]; c[i + 8] = hi[i]; }
    return __builtin_bit_cast(v16bf, c);
}

// ---------------- kernel 0: weight fp32 -> bf16 (runs once, lives in L2) ----
__global__ void __launch_bounds__(256)
wcvt_kernel(const float* __restrict__ w, unsigned short* __restrict__ wb, int n4) {
    int i = blockIdx.x * blockDim.x + threadIdx.x;
    if (i >= n4) return;
    f32x4 v = ((const f32x4*)w)[i];
    u16x4 h;
    h.x = f32_to_bf16_rne(v.x);
    h.y = f32_to_bf16_rne(v.y);
    h.z = f32_to_bf16_rne(v.z);
    h.w = f32_to_bf16_rne(v.w);
    ((u16x4*)wb)[i] = h;
}

// ---------------- kernel 1: fused router GEMM + group-limited top-k --------
__global__ void __launch_bounds__(256)
gate_kernel(const float* __restrict__ x,
            const unsigned short* __restrict__ wb,   // bf16 weight [E_DIM][D_DIM]
            const float* __restrict__ bias,
            float* __restrict__ outw,                // [N][TOPK]
            int* __restrict__ outi)                  // [N][TOPK]
{
    __shared__ unsigned short sA[2][ROWS_PER_BLOCK * LDS_STRIDE];  // double buffer
    __shared__ float sScores[ROWS_PER_BLOCK][E_DIM + 4];           // bank-skewed
    __shared__ float sBias[E_DIM];

    const int tid  = threadIdx.x;
    const int wave = tid >> 5;          // 0..7  == routing group id
    const int lane = tid & 31;
    const int l    = lane & 15;         // M (for A) / N (for B) within tile
    const int g    = lane >> 4;         // lane-group -> K half selector
    const int rowbase = blockIdx.x * ROWS_PER_BLOCK;

    sBias[tid] = bias[tid];             // blockDim == E_DIM == 256

    const v8f vzero = {0.f,0.f,0.f,0.f,0.f,0.f,0.f,0.f};
    v8f acc[MTILES][2];
#pragma unroll
    for (int t = 0; t < MTILES; ++t)
#pragma unroll
        for (int u = 0; u < 2; ++u) acc[t][u] = vzero;

    const int eb0 = wave * 32;          // this wave's 32 experts (== its group)

    // staging coordinates for this thread (KC/4 = 32 quads per row)
    const int srow0 = tid >> 5;         // first row this thread touches (8 rows/sweep)
    const int scq   = tid & 31;         // quad-of-4-floats within row

    // ---- prologue: prefetch chunk 0 into registers ----
    f32x4 stage[PF];
#pragma unroll
    for (int it = 0; it < PF; ++it) {
        int row = srow0 + it * 8;
        stage[it] = *(const f32x4*)(x + (size_t)(rowbase + row) * D_DIM + scq * 4);
    }

    int p = 0;
    for (int kc = 0; kc < D_DIM; kc += KC) {
        // ---- convert + store prefetched chunk into LDS buffer p ----
#pragma unroll
        for (int it = 0; it < PF; ++it) {
            int row = srow0 + it * 8;
            u16x4 h;
            h.x = f32_to_bf16_rne(stage[it].x);
            h.y = f32_to_bf16_rne(stage[it].y);
            h.z = f32_to_bf16_rne(stage[it].z);
            h.w = f32_to_bf16_rne(stage[it].w);
            *(u16x4*)(&sA[p][row * LDS_STRIDE + scq * 4]) = h;
        }
        __syncthreads();                // single barrier per chunk (2-buffer safe)

        // ---- issue next chunk's global loads (overlap with WMMA below) ----
        if (kc + KC < D_DIM) {
#pragma unroll
            for (int it = 0; it < PF; ++it) {
                int row = srow0 + it * 8;
                stage[it] = *(const f32x4*)(x + (size_t)(rowbase + row) * D_DIM
                                            + (kc + KC) + scq * 4);
            }
        }

        // ---- k-loop over this chunk: 4 steps x 8 WMMA ----
#pragma unroll 2
        for (int kk = 0; kk < KC; kk += 32) {
            v16bf bfr[2];
#pragma unroll
            for (int u = 0; u < 2; ++u) {
                // B 32x16 layout (same k order as A): lane-group g holds contiguous
                // K in [16g, 16g+16) for column eb0+u*16+l -> 32B straight read
                const unsigned short* bp =
                    wb + (size_t)(eb0 + u * 16 + l) * D_DIM + kc + kk + 16 * g;
                u16x8 lo = *(const u16x8*)bp;
                u16x8 hi = *(const u16x8*)(bp + 8);
                bfr[u] = combine16(lo, hi);
            }
#pragma unroll
            for (int t = 0; t < MTILES; ++t) {
                // A 16x32 layout (matches B): lane-group g holds contiguous
                // K in [16g, 16g+16)
                const unsigned short* ap =
                    &sA[p][(t * 16 + l) * LDS_STRIDE + kk + 16 * g];
                u16x8 lo = *(const u16x8*)ap;
                u16x8 hi = *(const u16x8*)(ap + 8);
                v16bf afr = combine16(lo, hi);
#pragma unroll
                for (int u = 0; u < 2; ++u)
                    acc[t][u] = __builtin_amdgcn_wmma_f32_16x16x32_bf16(
                        false, afr, false, bfr[u],
                        (short)0, acc[t][u], false, false);
            }
        }
        p ^= 1;
    }

    // ---- epilogue: sigmoid (fast rcp) -> LDS score tile ----
    // C layout: vgpr r, lane-group h: (M = r + 8h, N = l)
#pragma unroll
    for (int t = 0; t < MTILES; ++t)
#pragma unroll
        for (int u = 0; u < 2; ++u)
#pragma unroll
            for (int r = 0; r < 8; ++r) {
                int row = t * 16 + r + 8 * g;
                int e   = eb0 + u * 16 + l;
                float logit = acc[t][u][r];
                sScores[row][e] = __builtin_amdgcn_rcpf(1.0f + __expf(-logit));
            }
    __syncthreads();

    // ---- routing: one lane per row (2 waves active) ----
    if (tid < ROWS_PER_BLOCK) {
        const int row = tid;
        const float NEG = -3.0e38f;

        // group score = sum of top-2 (score + bias) per group of 32
        float gsum[NGRP];
#pragma unroll
        for (int grp = 0; grp < NGRP; ++grp) {
            float m1 = NEG, m2 = NEG;
            for (int e = 0; e < 32; ++e) {
                int   ei = grp * 32 + e;
                float v  = sScores[row][ei] + sBias[ei];
                if (v > m1)      { m2 = m1; m1 = v; }
                else if (v > m2) { m2 = v; }
            }
            gsum[grp] = m1 + m2;
        }

        // top-4 groups
        unsigned keep = 0u;
#pragma unroll
        for (int s = 0; s < TOPG; ++s) {
            int bi = 0; float bv = NEG;
            for (int grp = 0; grp < NGRP; ++grp)
                if (!((keep >> grp) & 1u) && gsum[grp] > bv) { bv = gsum[grp]; bi = grp; }
            keep |= 1u << bi;
        }

        // top-8 experts among kept groups (on score+bias), weights from raw score
        unsigned chosen[NGRP] = {0,0,0,0,0,0,0,0};
        int   idx[TOPK];
        float wsel[TOPK];
        float sum = 0.0f;
        for (int s = 0; s < TOPK; ++s) {
            int bi = 0; float bv = NEG;
            for (int grp = 0; grp < NGRP; ++grp) {
                if (!((keep >> grp) & 1u)) continue;
                for (int e = 0; e < 32; ++e) {
                    if ((chosen[grp] >> e) & 1u) continue;
                    int   ei = grp * 32 + e;
                    float v  = sScores[row][ei] + sBias[ei];
                    if (v > bv) { bv = v; bi = ei; }
                }
            }
            chosen[bi >> 5] |= 1u << (bi & 31);
            idx[s]  = bi;
            wsel[s] = sScores[row][bi];
            sum    += wsel[s];
        }

        float scale = ROUTE_SCALE * __builtin_amdgcn_rcpf(sum);
        size_t obase = (size_t)(rowbase + row) * TOPK;
#pragma unroll
        for (int s = 0; s < TOPK; ++s) {
            outw[obase + s] = wsel[s] * scale;
            outi[obase + s] = idx[s];
        }
    }
}

// ---------------- host launch ----------------
extern "C" void kernel_launch(void* const* d_in, const int* in_sizes, int n_in,
                              void* d_out, int out_size, void* d_ws, size_t ws_size,
                              hipStream_t stream) {
    const float* x    = (const float*)d_in[0];   // [N, D]
    const float* w    = (const float*)d_in[1];   // [E, D]
    const float* bias = (const float*)d_in[2];   // [E]
    const int N = in_sizes[0] / D_DIM;

    unsigned short* wb = (unsigned short*)d_ws;  // bf16 weight, 3.67 MB

    int n4 = (E_DIM * D_DIM) / 4;
    wcvt_kernel<<<(n4 + 255) / 256, 256, 0, stream>>>(w, wb, n4);

    float* outw = (float*)d_out;                  // first N*8 floats
    int*   outi = (int*)d_out + (size_t)N * TOPK; // then N*8 int32

    gate_kernel<<<N / ROWS_PER_BLOCK, 256, 0, stream>>>(x, wb, bias, outw, outi);
}
